// ParticleEncoder_83794811945081
// MI455X (gfx1250) — compile-verified, matrix operand order checked by reading, not run on hardware
//
#include <hip/hip_runtime.h>
#include <hip/hip_bf16.h>
#include <math.h>

// ---------------------------------------------------------------------------
// Types for CDNA5 WMMA (wave32): v_wmma_f32_16x16x32_bf16
// ---------------------------------------------------------------------------
typedef __attribute__((ext_vector_type(16))) __bf16 v16bf;
typedef __attribute__((ext_vector_type(8)))  float  v8f;

#define RADIUS_F 0.1125f
#define EPSF     1e-12f

static const int NF = 16000, NB = 8000, KF = 64, KB = 32;
static const int SLICE = 3200, NSL = 5;          // 3200 = 128 * 25
static const int KD0 = 256;                      // 64 bins * 4 (cin padded)
static const int KD1 = 6144 + 96;                // conv1 bins + dense1 tail
static const int KD2 = 4096 + 64;                // conv2 bins + dense2 tail

__device__ __forceinline__ unsigned short f2bf(float x) {
  unsigned u = __float_as_uint(x);
  unsigned r = u + 0x7FFFu + ((u >> 16) & 1u);   // round-to-nearest-even
  return (unsigned short)(r >> 16);
}

__device__ __forceinline__ void upd2(float2& d, float w, float2 f) {
  float2 t = d;
  t.x += w * f.x;
  t.y += w * f.y;
  d = t;                                          // ds_load_b64 + ds_store_b64
}

// ---------------------------------------------------------------------------
// 0) materialize feature sources: fluid [1,vel] -> [N,4]; box normals -> [N,4]
// ---------------------------------------------------------------------------
__global__ void k_init_feats(const float* __restrict__ vel,
                             const float* __restrict__ boxn,
                             float* __restrict__ ffeat, float* __restrict__ bfeat) {
  int t = blockIdx.x * 256 + threadIdx.x;
  if (t < NF) {
    ffeat[t * 4 + 0] = 1.0f;
    ffeat[t * 4 + 1] = vel[t * 3 + 0];
    ffeat[t * 4 + 2] = vel[t * 3 + 1];
    ffeat[t * 4 + 3] = vel[t * 3 + 2];
  }
  int b = t - NF;
  if (b >= 0 && b < NB) {
    bfeat[b * 4 + 0] = boxn[b * 3 + 0];
    bfeat[b * 4 + 1] = boxn[b * 3 + 1];
    bfeat[b * 4 + 2] = boxn[b * 3 + 2];
    bfeat[b * 4 + 3] = 0.0f;                      // cin pad -> zero contribution
  }
}

// ---------------------------------------------------------------------------
// 1) per-(query,neighbor) geometry: window * trilinear corner weights + binbase
// ---------------------------------------------------------------------------
__global__ void k_geom(const float* __restrict__ posq, const float* __restrict__ poss,
                       const int* __restrict__ idx, const unsigned char* __restrict__ mask,
                       float* __restrict__ wout, int* __restrict__ binout,
                       int Nq, int K) {
  int t = blockIdx.x * 256 + threadIdx.x;
  if (t >= Nq * K) return;
  int q = t / K;
  int j = idx[t];

  float dx = (poss[j * 3 + 0] - posq[q * 3 + 0]) * (1.0f / RADIUS_F);
  float dy = (poss[j * 3 + 1] - posq[q * 3 + 1]) * (1.0f / RADIUS_F);
  float dz = (poss[j * 3 + 2] - posq[q * 3 + 2]) * (1.0f / RADIUS_F);
  float sq = dx * dx + dy * dy + dz * dz;

  float om  = 1.0f - sq;
  float win = fminf(fmaxf(om * om * om, 0.0f), 1.0f);
  if (!mask[t]) win = 0.0f;

  // ---- ball -> cylinder ----
  float nrm   = sqrtf(sq);
  float xy_sq = dx * dx + dy * dy;
  bool  polar = (1.25f * dz * dz) > xy_sq;
  float s_pol = sqrtf(3.0f * nrm / (nrm + fabsf(dz) + EPSF));
  float s_eq  = nrm / sqrtf(xy_sq + EPSF);
  float s     = polar ? s_pol : s_eq;
  float xc = dx * s, yc = dy * s;
  float zc = polar ? copysignf(nrm, dz) : 1.5f * dz;
  if (sq < EPSF) { xc = 0.0f; yc = 0.0f; zc = 0.0f; }

  // ---- cylinder -> cube (inverse Shirley-Chiu) ----
  float rxy  = sqrtf(xc * xc + yc * yc);
  bool  xbig = fabsf(xc) >= fabsf(yc);
  float sx   = (fabsf(xc) > EPSF) ? xc : 1.0f;
  float sy   = (fabsf(yc) > EPSF) ? yc : 1.0f;
  const float FOP = 1.2732395447351627f;          // 4/pi
  float sgnx = (xc > 0.0f) ? 1.0f : ((xc < 0.0f) ? -1.0f : 0.0f);
  float sgny = (yc > 0.0f) ? 1.0f : ((yc < 0.0f) ? -1.0f : 0.0f);
  float u1 = sgnx * rxy;
  float v1 = u1 * FOP * atanf(yc / sx);
  float v2 = sgny * rxy;
  float u2 = v2 * FOP * atanf(xc / sy);
  float u = xbig ? u1 : u2;
  float v = xbig ? v1 : v2;
  if (rxy < EPSF) { u = 0.0f; v = 0.0f; }

  // ---- align_corners grid, ks=4 per axis ----
  float gx = (u  * 0.5f + 0.5f) * 3.0f;
  float gy = (v  * 0.5f + 0.5f) * 3.0f;
  float gz = (zc * 0.5f + 0.5f) * 3.0f;
  float fx = fminf(fmaxf(floorf(gx), 0.0f), 2.0f);
  float fy = fminf(fmaxf(floorf(gy), 0.0f), 2.0f);
  float fz = fminf(fmaxf(floorf(gz), 0.0f), 2.0f);
  int ix = (int)fx, iy = (int)fy, iz = (int)fz;
  float tx = gx - fx, ty = gy - fy, tz = gz - fz;

  binout[t] = (iz * 4 + iy) * 4 + ix;

  float x0 = (1.0f - tx) * win, x1 = tx * win;
  float y0 = 1.0f - ty, y1 = ty;
  float z0 = 1.0f - tz, z1 = tz;
  float4 wa, wb;                                  // corner cc = dz*4 + dy*2 + dx
  wa.x = z0 * y0 * x0; wa.y = z0 * y0 * x1; wa.z = z0 * y1 * x0; wa.w = z0 * y1 * x1;
  wb.x = z1 * y0 * x0; wb.y = z1 * y0 * x1; wb.z = z1 * y1 * x0; wb.w = z1 * y1 * x1;
  ((float4*)wout)[t * 2 + 0] = wa;
  ((float4*)wout)[t * 2 + 1] = wb;
}

// ---------------------------------------------------------------------------
// 2) wave-per-query scatter into LDS bins (f32, b64 RMWs), bf16 write-out
// ---------------------------------------------------------------------------
template <int CIN, int K, int TAIL, int QPB>
__global__ __launch_bounds__(QPB * 32)
void k_scatter(const float* __restrict__ wgt, const int* __restrict__ bins,
               const int* __restrict__ idx, const float* __restrict__ feat,
               int featStride, unsigned short* __restrict__ A, int Kdim,
               int rowOff, int rowCount) {
  __shared__ float sm[QPB][64 * CIN];
  int wave = threadIdx.x >> 5, lane = threadIdx.x & 31;
  int ql = blockIdx.x * QPB + wave;               // slice-local row
  if (ql >= rowCount) return;
  int q = rowOff + ql;                            // global query id
  float*  a  = sm[wave];                          // wave-private LDS region
  float2* a2 = (float2*)a;

  for (int i = lane; i < (64 * CIN) / 2; i += 32) a2[i] = make_float2(0.0f, 0.0f);

  const float4* wq = (const float4*)wgt + (size_t)q * K * 2;
  const int*    bq = bins + (size_t)q * K;
  const int*    iq = idx  + (size_t)q * K;

  for (int k = 0; k < K; ++k) {
    int j  = iq[k];
    int bb = bq[k];
    float4 wa = wq[k * 2 + 0];
    float4 wb = wq[k * 2 + 1];
    const float* fj = feat + (size_t)j * featStride;

    // paired channels: 64 channels per pass, b64 LDS RMWs
    #pragma unroll
    for (int ch = 0; ch < CIN / 64; ++ch) {
      int c2 = ch * 32 + lane;                    // float2 index within row
      float2 f = ((const float2*)fj)[c2];
      int b2 = bb * (CIN / 2) + c2;               // bin offsets {0,1,4,5,16,17,20,21}
      upd2(a2[b2 +  0 * (CIN / 2)], wa.x, f);
      upd2(a2[b2 +  1 * (CIN / 2)], wa.y, f);
      upd2(a2[b2 +  4 * (CIN / 2)], wa.z, f);
      upd2(a2[b2 +  5 * (CIN / 2)], wa.w, f);
      upd2(a2[b2 + 16 * (CIN / 2)], wb.x, f);
      upd2(a2[b2 + 17 * (CIN / 2)], wb.y, f);
      upd2(a2[b2 + 20 * (CIN / 2)], wb.z, f);
      upd2(a2[b2 + 21 * (CIN / 2)], wb.w, f);
    }
    // remainder (<64 channels): scalar b32 RMWs
    if (CIN % 64) {
      int c = (CIN / 64) * 64 + lane;
      if (c < CIN) {
        float f = fj[c];
        int base = bb * CIN + c;
        a[base +  0 * CIN] += wa.x * f;
        a[base +  1 * CIN] += wa.y * f;
        a[base +  4 * CIN] += wa.z * f;
        a[base +  5 * CIN] += wa.w * f;
        a[base + 16 * CIN] += wb.x * f;
        a[base + 17 * CIN] += wb.y * f;
        a[base + 20 * CIN] += wb.z * f;
        a[base + 21 * CIN] += wb.w * f;
      }
    }
  }

  // bf16 write-out of the bin block (4 values per store)
  uint2* out = (uint2*)(A + (size_t)ql * Kdim);
  for (int i = lane; i < (64 * CIN) / 4; i += 32) {
    float2 p = a2[2 * i + 0];
    float2 r = a2[2 * i + 1];
    uint2 u;
    u.x = (unsigned)f2bf(p.x) | ((unsigned)f2bf(p.y) << 16);
    u.y = (unsigned)f2bf(r.x) | ((unsigned)f2bf(r.y) << 16);
    out[i] = u;
  }

  // dense-skip tail: raw features appended as extra K rows
  if (TAIL > 0) {
    const float* fq = feat + (size_t)q * featStride;
    for (int c = lane; c < TAIL; c += 32)
      A[(size_t)ql * Kdim + 64 * CIN + c] = f2bf(fq[c]);
  }
}

// ---------------------------------------------------------------------------
// 3) pre-shuffle filter into per-lane B-fragment layout (bf16), pad/append rows
//    B (32x16) lane map mirrors documented A map: n = nt*16 + lane%16,
//    k = kc*32 + (lane<16?0:8) + (v<4?0:16) + 2*(v&3) (+1 in high half)
// ---------------------------------------------------------------------------
__global__ void k_prep_frag(const float* __restrict__ Fmain, int mainRowsDst,
                            int cinSrc, int cinDst,
                            const float* __restrict__ Ftail, int tailRows, int NC,
                            unsigned* __restrict__ dst, int kchunks, int ntiles) {
  int t = blockIdx.x * 256 + threadIdx.x;
  int total = kchunks * ntiles * 32 * 8;
  if (t >= total) return;
  int v    = t & 7;
  int lane = (t >> 3) & 31;
  int rest = t >> 8;
  int nt   = rest % ntiles;
  int kc   = rest / ntiles;
  int n  = nt * 16 + (lane & 15);
  int k0 = kc * 32 + ((lane < 16) ? 0 : 8) + ((v < 4) ? 0 : 16) + 2 * (v & 3);

  float val[2];
  #pragma unroll
  for (int h = 0; h < 2; ++h) {
    int k = k0 + h;
    float x = 0.0f;
    if (k < mainRowsDst) {
      int bin = k / cinDst, ci = k - bin * cinDst;
      if (ci < cinSrc) x = Fmain[((size_t)bin * cinSrc + ci) * NC + n];
    } else {
      int tr = k - mainRowsDst;
      if (Ftail && tr < tailRows) x = Ftail[(size_t)tr * NC + n];
    }
    val[h] = x;
  }
  dst[t] = (unsigned)f2bf(val[0]) | ((unsigned)f2bf(val[1]) << 16);
}

// ---------------------------------------------------------------------------
// 4) bf16 WMMA GEMM, 2 M-tiles x NTILES N-tiles per wave (B-frag reuse, 2 ILP
//    chains). Block = 4 waves = 128 rows. Fused epilogues:
//    EPI 0: raw store   EPI 1: +bias1+bias2 -> out1, relu -> out2
//    EPI 2: +bias1+bias2+resid -> out1
// ---------------------------------------------------------------------------
template <int KCHUNKS, int NTILES, int EPI>
__global__ __launch_bounds__(128)
void k_gemm(const unsigned short* __restrict__ A, const unsigned* __restrict__ Bfrag,
            const float* __restrict__ bias1, const float* __restrict__ bias2,
            const float* __restrict__ resid,
            float* __restrict__ out1, float* __restrict__ out2, int rowOff) {
  constexpr int Kdim = KCHUNKS * 32;
  constexpr int NC   = NTILES * 16;
  int wave = threadIdx.x >> 5, lane = threadIdx.x & 31;
  int rowbase = blockIdx.x * 128 + wave * 32;     // slice-local, 32 rows per wave
  int m     = lane & 15;
  int khalf = (lane < 16) ? 0 : 8;
  const unsigned short* arow0 = A + (size_t)(rowbase + m) * Kdim + khalf;
  const unsigned short* arow1 = arow0 + (size_t)16 * Kdim;

  v8f acc[2][NTILES];
  #pragma unroll
  for (int mt = 0; mt < 2; ++mt)
    #pragma unroll
    for (int nt = 0; nt < NTILES; ++nt) acc[mt][nt] = (v8f){0, 0, 0, 0, 0, 0, 0, 0};

  for (int kc = 0; kc < KCHUNKS; ++kc) {
    union { uint4 u[2]; v16bf v; } af0, af1;
    const uint4* ap0 = (const uint4*)(arow0 + kc * 32);
    const uint4* ap1 = (const uint4*)(arow1 + kc * 32);
    af0.u[0] = ap0[0];                             // K khalf .. khalf+7
    af0.u[1] = ap0[2];                             // K khalf+16 .. khalf+23
    af1.u[0] = ap1[0];
    af1.u[1] = ap1[2];
    #pragma unroll
    for (int nt = 0; nt < NTILES; ++nt) {
      union { uint4 u[2]; v16bf v; } bfr;
      const uint4* bp = (const uint4*)Bfrag + ((size_t)((kc * NTILES + nt) * 32 + lane)) * 2;
      bfr.u[0] = bp[0];
      bfr.u[1] = bp[1];
      acc[0][nt] = __builtin_amdgcn_wmma_f32_16x16x32_bf16(
          false, af0.v, false, bfr.v, (short)0, acc[0][nt], false, false);
      acc[1][nt] = __builtin_amdgcn_wmma_f32_16x16x32_bf16(
          false, af1.v, false, bfr.v, (short)0, acc[1][nt], false, false);
    }
  }

  int moff = (lane >> 4) * 8;
  int nloc = lane & 15;
  #pragma unroll
  for (int mt = 0; mt < 2; ++mt) {
    #pragma unroll
    for (int nt = 0; nt < NTILES; ++nt) {
      int col = nt * 16 + nloc;
      float b = (EPI >= 1) ? (bias1[col] + bias2[col]) : 0.0f;
      #pragma unroll
      for (int r = 0; r < 8; ++r) {
        size_t grow = (size_t)(rowOff + rowbase + mt * 16 + moff + r);
        float vv = acc[mt][nt][r] + b;
        if (EPI == 2) vv += resid[grow * NC + col];
        out1[grow * NC + col] = vv;
        if (EPI == 1) out2[grow * NC + col] = fmaxf(vv, 0.0f);
      }
    }
  }
}

// ---------------------------------------------------------------------------
// 5) stage-0 combine: feats = [a_co+b, a_cf+b, dense0], inp1 = relu(feats)
// ---------------------------------------------------------------------------
__global__ void k_combine0(const float* __restrict__ aco, const float* __restrict__ acf,
                           const float* __restrict__ ffeat,
                           const float* __restrict__ b_co, const float* __restrict__ b_cf,
                           const float* __restrict__ wd0, const float* __restrict__ bd0,
                           float* __restrict__ inp1) {
  int t = blockIdx.x * 256 + threadIdx.x;
  if (t >= NF * 96) return;
  int n = t / 96, c = t - n * 96;
  float v;
  if (c < 32) {
    v = aco[n * 32 + c] + b_co[c];
  } else if (c < 64) {
    int cc = c - 32;
    v = acf[n * 32 + cc] + b_cf[cc];
  } else {
    int cc = c - 64;
    const float* f = ffeat + n * 4;
    v = f[0] * wd0[0 * 32 + cc] + f[1] * wd0[1 * 32 + cc] +
        f[2] * wd0[2 * 32 + cc] + f[3] * wd0[3 * 32 + cc] + bd0[cc];
  }
  inp1[t] = fmaxf(v, 0.0f);
}

// ---------------------------------------------------------------------------
// host-side launcher
// ---------------------------------------------------------------------------
extern "C" void kernel_launch(void* const* d_in, const int* in_sizes, int n_in,
                              void* d_out, int out_size, void* d_ws, size_t ws_size,
                              hipStream_t stream) {
  const float* pos  = (const float*)d_in[0];
  const float* vel  = (const float*)d_in[1];
  const float* box  = (const float*)d_in[2];
  const float* boxn = (const float*)d_in[3];
  const float* wc0f = (const float*)d_in[4];
  const float* bc0f = (const float*)d_in[5];
  const float* wc0o = (const float*)d_in[6];
  const float* bc0o = (const float*)d_in[7];
  const float* wd0  = (const float*)d_in[8];
  const float* bd0  = (const float*)d_in[9];
  const float* wc1  = (const float*)d_in[10];
  const float* bc1  = (const float*)d_in[11];
  const float* wd1  = (const float*)d_in[12];
  const float* bd1  = (const float*)d_in[13];
  const float* wc2  = (const float*)d_in[14];
  const float* bc2  = (const float*)d_in[15];
  const float* wd2  = (const float*)d_in[16];
  const float* bd2  = (const float*)d_in[17];
  const int*           nfi = (const int*)d_in[18];
  const unsigned char* nfm = (const unsigned char*)d_in[19];
  const int*           nbi = (const int*)d_in[20];
  const unsigned char* nbm = (const unsigned char*)d_in[21];
  float* out = (float*)d_out;

  // ---- workspace carve-up (256B aligned) ----
  size_t o = 0;
  char* ws = (char*)d_ws;
  auto alloc = [&](size_t bytes) -> char* {
    char* p = ws + o;
    o += (bytes + 255) & ~(size_t)255;
    return p;
  };
  float*          fgw  = (float*)alloc((size_t)NF * KF * 8 * 4);
  int*            fgb  = (int*)  alloc((size_t)NF * KF * 4);
  float*          bgw  = (float*)alloc((size_t)NF * KB * 8 * 4);
  int*            bgb  = (int*)  alloc((size_t)NF * KB * 4);
  float*          ff   = (float*)alloc((size_t)NF * 4 * 4);
  float*          bfeat= (float*)alloc((size_t)NB * 4 * 4);
  unsigned short* A0f  = (unsigned short*)alloc((size_t)NF * KD0 * 2);
  unsigned short* A0o  = (unsigned short*)alloc((size_t)NF * KD0 * 2);
  float*          acf  = (float*)alloc((size_t)NF * 32 * 4);
  float*          aco  = (float*)alloc((size_t)NF * 32 * 4);
  float*          inp1 = (float*)alloc((size_t)NF * 96 * 4);
  float*          ans1 = (float*)alloc((size_t)NF * 64 * 4);
  float*          inp2 = (float*)alloc((size_t)NF * 64 * 4);
  unsigned*       F0f  = (unsigned*)alloc((size_t)(KD0 / 32) * 2 * 32 * 32);
  unsigned*       F0o  = (unsigned*)alloc((size_t)(KD0 / 32) * 2 * 32 * 32);
  unsigned*       F1   = (unsigned*)alloc((size_t)(KD1 / 32) * 4 * 32 * 32);
  unsigned*       F2   = (unsigned*)alloc((size_t)(KD2 / 32) * 4 * 32 * 32);
  unsigned short* Abig = (unsigned short*)alloc((size_t)SLICE * KD1 * 2);
  (void)ws_size; (void)in_sizes; (void)n_in; (void)out_size;

  // ---- 0: features ----
  k_init_feats<<<(NF + NB + 255) / 256, 256, 0, stream>>>(vel, boxn, ff, bfeat);

  // ---- 1: geometry ----
  k_geom<<<(NF * KF + 255) / 256, 256, 0, stream>>>(pos, pos, nfi, nfm, fgw, fgb, NF, KF);
  k_geom<<<(NF * KB + 255) / 256, 256, 0, stream>>>(pos, box, nbi, nbm, bgw, bgb, NF, KB);

  // ---- 2: filter fragments ----
  k_prep_frag<<<(8 * 2 * 32 * 8 + 255) / 256, 256, 0, stream>>>(
      wc0f, 256, 4, 4, nullptr, 0, 32, F0f, KD0 / 32, 2);
  k_prep_frag<<<(8 * 2 * 32 * 8 + 255) / 256, 256, 0, stream>>>(
      wc0o, 256, 3, 4, nullptr, 0, 32, F0o, KD0 / 32, 2);
  k_prep_frag<<<((KD1 / 32) * 4 * 32 * 8 + 255) / 256, 256, 0, stream>>>(
      wc1, 6144, 96, 96, wd1, 96, 64, F1, KD1 / 32, 4);
  k_prep_frag<<<((KD2 / 32) * 4 * 32 * 8 + 255) / 256, 256, 0, stream>>>(
      wc2, 4096, 64, 64, wd2, 64, 64, F2, KD2 / 32, 4);

  // ---- 3: layer 0 (cin=4 padded), full 16000 rows ----
  k_scatter<4, KF, 0, 8><<<NF / 8, 256, 0, stream>>>(fgw, fgb, nfi, ff, 4, A0f, KD0, 0, NF);
  k_scatter<4, KB, 0, 8><<<NF / 8, 256, 0, stream>>>(bgw, bgb, nbi, bfeat, 4, A0o, KD0, 0, NF);
  k_gemm<KD0 / 32, 2, 0><<<NF / 128, 128, 0, stream>>>(A0f, F0f, nullptr, nullptr, nullptr,
                                                       acf, nullptr, 0);
  k_gemm<KD0 / 32, 2, 0><<<NF / 128, 128, 0, stream>>>(A0o, F0o, nullptr, nullptr, nullptr,
                                                       aco, nullptr, 0);
  k_combine0<<<(NF * 96 + 255) / 256, 256, 0, stream>>>(aco, acf, ff, bc0o, bc0f, wd0, bd0, inp1);

  // ---- 4: layer 1 (cin=96, dense tail 96), sliced to bound workspace ----
  for (int s = 0; s < NSL; ++s) {
    int ro = s * SLICE;
    k_scatter<96, KF, 96, 2><<<SLICE / 2, 64, 0, stream>>>(fgw, fgb, nfi, inp1, 96,
                                                           Abig, KD1, ro, SLICE);
    k_gemm<KD1 / 32, 4, 1><<<SLICE / 128, 128, 0, stream>>>(Abig, F1, bc1, bd1, nullptr,
                                                            ans1, inp2, ro);
  }

  // ---- 5: layer 2 (cin=64, dense tail 64, residual ans1) -> d_out ----
  for (int s = 0; s < NSL; ++s) {
    int ro = s * SLICE;
    k_scatter<64, KF, 64, 2><<<SLICE / 2, 64, 0, stream>>>(fgw, fgb, nfi, inp2, 64,
                                                           Abig, KD2, ro, SLICE);
    k_gemm<KD2 / 32, 4, 2><<<SLICE / 128, 128, 0, stream>>>(Abig, F2, bc2, bd2, ans1,
                                                            out, nullptr, ro);
  }
}
